// MBDDegradationImputation_38208029065953
// MI455X (gfx1250) — matrix operand order repaired and checked by reading, not compiled, not run on hardware
//
#include <hip/hip_runtime.h>
#include <hip/hip_bf16.h>
#include <math.h>

#define N_SAMP 2048
#define T_DIM  512
#define F_DIM  64
#define TF     32768   // T_DIM * F_DIM
#define NPART  4       // n-dimension split for the weighted pass
#define NCHUNK (N_SAMP / NPART)   // 512 samples per chunk -> 16 WMMA steps

typedef _Float16 v16h __attribute__((ext_vector_type(16)));
typedef float    v8f  __attribute__((ext_vector_type(8)));

// ---------------------------------------------------------------------------
// K0: cosine-beta schedule scalars (f64, matches reference), 1 thread.
// ws[0] = sigma_i = sqrt(1 - acp[i]) ; ws[1] = c1 = sqrt(abar_i)/(sqrt(alpha_i)*sqrt(abar_{i-1}))
// ---------------------------------------------------------------------------
__device__ __forceinline__ double ac_of(double x) {
    const double PI_H = 1.5707963267948966; // pi/2
    double c = cos((x / 1000.0 + 0.008) / 1.008 * PI_H);
    return c * c; // normalization by ac[0] cancels in the ratio
}

__global__ void k_sched(const int* __restrict__ ip, float* __restrict__ ws) {
    const int i = ip[0];
    double prod = 1.0, alpha_i = 1.0, acp_im1 = 1.0, acp_i = 1.0;
    double prev = ac_of(0.0);
    for (int j = 0; j <= i; ++j) {
        double cur  = ac_of((double)(j + 1));
        double beta = 1.0 - cur / prev;
        beta = beta < 0.0 ? 0.0 : (beta > 0.999 ? 0.999 : beta);
        double alpha = 1.0 - beta;
        prod *= alpha;
        if (j == i - 1) acp_im1 = prod;
        if (j == i) { alpha_i = alpha; acp_i = prod; }
        prev = cur;
    }
    float ab   = (float)acp_i;
    float abm1 = (float)acp_im1;
    float al   = (float)alpha_i;
    ws[0] = (float)sqrt(1.0 - acp_i);                       // sigma_i
    ws[1] = sqrtf(ab) / (sqrtf(al) * sqrtf(abm1));          // c1 (~= 1.0f)
}

// ---------------------------------------------------------------------------
// K1: scores[n] = -mean((cand - pred)^2), one block per n. Streams 128 KB/row
// with b128 loads + global_prefetch; aux arrays (Xbar/obs 128 KB, mask 32 KB) stay L2-hot.
// ---------------------------------------------------------------------------
__global__ void __launch_bounds__(256)
k_scores(const float* __restrict__ eps, const float* __restrict__ Xb,
         const float* __restrict__ obs, const unsigned char* __restrict__ mask,
         const float* __restrict__ tp,  const float* __restrict__ da,
         const float* __restrict__ db,  const float* __restrict__ wsS,
         float* __restrict__ scores) {
    const int   n     = blockIdx.x;
    const int   tid   = threadIdx.x;
    const float sigma = wsS[0];
    const float* e = eps + (size_t)n * TF;
    float acc = 0.0f;
    for (int it = 0; it < TF / (256 * 4); ++it) {
        const int q = (it * 256 + tid) * 4;            // element index, mult of 4
        __builtin_prefetch(e + q + 4096, 0, 0);        // global_prefetch_b8, 16KB ahead
        const float4 e4 = *reinterpret_cast<const float4*>(e + q);
        const float4 x4 = *reinterpret_cast<const float4*>(Xb + q);
        const float4 o4 = *reinterpret_cast<const float4*>(obs + q);
        const unsigned m4 = *reinterpret_cast<const unsigned*>(mask + q);
        const float  tpv = tp[q >> 6];
        const float4 a4 = *reinterpret_cast<const float4*>(da + (q & 63));
        const float4 b4 = *reinterpret_cast<const float4*>(db + (q & 63));
        {
            float pr = fmaf(b4.x, tpv, a4.x);
            float x0 = fminf(fmaxf(fmaf(e4.x, sigma, x4.x), -1.0f), 1.0f);
            float r  = (m4 & 0xFFu) ? (o4.x - pr) : (x0 - pr);
            acc = fmaf(r, r, acc);
        }
        {
            float pr = fmaf(b4.y, tpv, a4.y);
            float x0 = fminf(fmaxf(fmaf(e4.y, sigma, x4.y), -1.0f), 1.0f);
            float r  = (m4 & 0xFF00u) ? (o4.y - pr) : (x0 - pr);
            acc = fmaf(r, r, acc);
        }
        {
            float pr = fmaf(b4.z, tpv, a4.z);
            float x0 = fminf(fmaxf(fmaf(e4.z, sigma, x4.z), -1.0f), 1.0f);
            float r  = (m4 & 0xFF0000u) ? (o4.z - pr) : (x0 - pr);
            acc = fmaf(r, r, acc);
        }
        {
            float pr = fmaf(b4.w, tpv, a4.w);
            float x0 = fminf(fmaxf(fmaf(e4.w, sigma, x4.w), -1.0f), 1.0f);
            float r  = (m4 & 0xFF000000u) ? (o4.w - pr) : (x0 - pr);
            acc = fmaf(r, r, acc);
        }
    }
    __shared__ float red[256];
    red[tid] = acc;
    __syncthreads();
    for (int st = 128; st > 0; st >>= 1) {
        if (tid < st) red[tid] += red[tid + st];
        __syncthreads();
    }
    if (tid == 0) scores[n] = -red[0] * (1.0f / (float)TF);
}

// ---------------------------------------------------------------------------
// K2: softmax((s - mean)/std(ddof=1)/0.1) -> f16 weights. Single block.
// ---------------------------------------------------------------------------
__global__ void __launch_bounds__(256)
k_weights(const float* __restrict__ scores, _Float16* __restrict__ wh) {
    __shared__ float red[256];
    const int tid = threadIdx.x;
    float s[8];
    float part = 0.0f;
    for (int j = 0; j < 8; ++j) { s[j] = scores[tid + 256 * j]; part += s[j]; }
    red[tid] = part; __syncthreads();
    for (int st = 128; st > 0; st >>= 1) { if (tid < st) red[tid] += red[tid + st]; __syncthreads(); }
    const float mean = red[0] * (1.0f / (float)N_SAMP);
    __syncthreads();

    part = 0.0f;
    for (int j = 0; j < 8; ++j) { float d = s[j] - mean; part = fmaf(d, d, part); }
    red[tid] = part; __syncthreads();
    for (int st = 128; st > 0; st >>= 1) { if (tid < st) red[tid] += red[tid + st]; __syncthreads(); }
    const float stdv  = fmaxf(sqrtf(red[0] / (float)(N_SAMP - 1)), 1e-4f);
    const float scale = 1.0f / (stdv * 0.1f);
    __syncthreads();

    float l[8];
    float lm = -3.4e38f;
    for (int j = 0; j < 8; ++j) { l[j] = (s[j] - mean) * scale; lm = fmaxf(lm, l[j]); }
    red[tid] = lm; __syncthreads();
    for (int st = 128; st > 0; st >>= 1) { if (tid < st) red[tid] = fmaxf(red[tid], red[tid + st]); __syncthreads(); }
    const float lmax = red[0];
    __syncthreads();

    float e[8];
    part = 0.0f;
    for (int j = 0; j < 8; ++j) { e[j] = expf(l[j] - lmax); part += e[j]; }
    red[tid] = part; __syncthreads();
    for (int st = 128; st > 0; st >>= 1) { if (tid < st) red[tid] += red[tid + st]; __syncthreads(); }
    const float invZ = 1.0f / red[0];
    for (int j = 0; j < 8; ++j) wh[tid + 256 * j] = (_Float16)(e[j] * invZ);
}

// ---------------------------------------------------------------------------
// K3: partial weighted sums over an n-chunk via v_wmma_f32_16x16x32_f16.
// wave w: p-tile = w & 2047 (16 outputs), n-chunk = w >> 11 (512 samples,
// 16 WMMA steps). eps loads are double-buffered in registers so the wait
// before each WMMA covers loads issued one full convert+WMMA earlier.
//   B[k][j] = f16(clip(eps[n0+k, p0+j]*sigma + Xbar[p0+j]))   (k = 0..31)
//   A[m][k] = f16 weight w[n0+k]  (rows replicated -> D rows identical)
// A and B both use contiguous k = 16h + s; partial D row 0 stored to ws (no float
// atomics -> bitwise-deterministic across replays).
// ---------------------------------------------------------------------------
__global__ void __launch_bounds__(256)
k_weighted(const float* __restrict__ eps, const float* __restrict__ Xb,
           const float* __restrict__ wsS, const _Float16* __restrict__ wh,
           float* __restrict__ partials) {
    const int wave  = (int)((blockIdx.x * blockDim.x + threadIdx.x) >> 5); // 0..8191
    const int lane  = threadIdx.x & 31;
    const int tile  = wave & (TF / 16 - 1);   // 0..2047
    const int chunk = wave >> 11;             // 0..3
    const int p     = tile * 16 + (lane & 15);
    const int kb    = (lane >> 4) << 4;   // B: K base 0 or 16 (half-wave split)
    const int ka    = (lane >> 4) << 3;   // unused; A also indexes with kb (+16 K)
    const int nbeg  = chunk * NCHUNK;
    const int nend  = nbeg + NCHUNK;
    const float sigma = wsS[0];
    const float xb    = Xb[p];

    // prologue: load first 32-deep slab of eps for this tile
    float ev[16];
    {
        const float* ep0 = eps + (size_t)(nbeg + kb) * TF + p;
#pragma unroll
        for (int k = 0; k < 16; ++k) ev[k] = ep0[(size_t)k * TF];
    }

    v8f c = {};
    for (int n0 = nbeg; n0 < nend; n0 += 32) {
        // convert current slab to f16 B operand
        v16h b;
#pragma unroll
        for (int k = 0; k < 16; ++k) {
            float x0 = fminf(fmaxf(fmaf(ev[k], sigma, xb), -1.0f), 1.0f);
            b[k] = (_Float16)x0;
        }
        // issue next slab's loads before the WMMA (register double-buffer)
        if (n0 + 32 < nend) {
            const float* epn = eps + (size_t)(n0 + 32 + kb) * TF + p;
#pragma unroll
            for (int k = 0; k < 16; ++k) ev[k] = epn[(size_t)k * TF];
        }
        // broadcast weights into A (replicated rows)
        v16h a;
#pragma unroll
        for (int k = 0; k < 8; ++k) {
            a[k]     = wh[n0 + kb + k];        // K = 0..7   (or 16..23)
            a[k + 8] = wh[n0 + kb + 8 + k];    // K = 8..15  (or 24..31)
        }
        c = __builtin_amdgcn_wmma_f32_16x16x32_f16(
                /*neg_a=*/false, a, /*neg_b=*/false, b,
                /*c_mod=*/(short)0, c, /*reuse_a=*/false, /*reuse_b=*/false);
    }
    if (lane < 16) // D row M=0: element 0 of lanes 0..15, col N = lane
        partials[(size_t)chunk * TF + p] = c[0];
}

// ---------------------------------------------------------------------------
// K4: combine partials + fused epilogue: out = mask ? obs : c1 * sum(partials)
// ---------------------------------------------------------------------------
__global__ void __launch_bounds__(256)
k_combine(const float* __restrict__ partials, const float* __restrict__ obs,
          const unsigned char* __restrict__ mask, const float* __restrict__ wsS,
          float* __restrict__ out) {
    const int p = blockIdx.x * 256 + threadIdx.x;
    const float c1 = wsS[1];
    float s = partials[p];
#pragma unroll
    for (int j = 1; j < NPART; ++j) s += partials[(size_t)j * TF + p];
    out[p] = mask[p] ? obs[p] : (c1 * s);
}

// ---------------------------------------------------------------------------
extern "C" void kernel_launch(void* const* d_in, const int* in_sizes, int n_in,
                              void* d_out, int out_size, void* d_ws, size_t ws_size,
                              hipStream_t stream) {
    const float*         Xbar = (const float*)d_in[0];
    const float*         obs  = (const float*)d_in[1];
    const float*         tp   = (const float*)d_in[2];
    const unsigned char* mask = (const unsigned char*)d_in[3];
    const float*         eps  = (const float*)d_in[4];
    const float*         da   = (const float*)d_in[5];
    const float*         db   = (const float*)d_in[6];
    const int*           ip   = (const int*)d_in[7];
    float* outp = (float*)d_out;

    float*     ws       = (float*)d_ws;
    float*     scores   = ws + 16;                              // 2048 f32
    _Float16*  wh       = (_Float16*)(ws + 16 + N_SAMP);        // 2048 f16
    float*     partials = ws + 16 + N_SAMP + N_SAMP / 2;        // NPART*TF f32

    k_sched   <<<1, 1, 0, stream>>>(ip, ws);
    k_scores  <<<N_SAMP, 256, 0, stream>>>(eps, Xbar, obs, mask, tp, da, db, ws, scores);
    k_weights <<<1, 256, 0, stream>>>(scores, wh);
    k_weighted<<<(NPART * TF / 16) / 8, 256, 0, stream>>>(eps, Xbar, ws, wh, partials);
    k_combine <<<TF / 256, 256, 0, stream>>>(partials, obs, mask, ws, outp);
    (void)in_sizes; (void)n_in; (void)out_size; (void)ws_size;
}
